// QuantizedLinear_44109314130114
// MI455X (gfx1250) — matrix operand order repaired and checked by reading, not compiled, not run on hardware
//
#include <hip/hip_runtime.h>
#include <hip/hip_bf16.h>
#include <math.h>

// ---------------------------------------------------------------------------
// QuantizedLinear for MI455X (gfx1250, wave32, WMMA)
//   Kernel 1: projected = X @ U via split-bf16 (hi/lo) on
//             v_wmma_f32_16x16x32_bf16 (3 products, ~1e-4 rel accuracy so the
//             codebook argmin decisions match fp32), software-pipelined
//             double-buffered LDS, fused two-level quantization epilogue,
//             writes A = (z*lx + r*dx)*lw as bf16 to workspace.
//   Kernel 2: out = A_bf16 @ bf16(Z_w) + bias (v_wmma_f32_16x16x32_bf16,
//             f32 accumulate), same pipelined structure.
// ---------------------------------------------------------------------------

typedef float v8f  __attribute__((ext_vector_type(8)));
typedef __bf16 bf16x4  __attribute__((ext_vector_type(4)));
typedef __bf16 bf16x8  __attribute__((ext_vector_type(8)));
typedef __bf16 bf16x16 __attribute__((ext_vector_type(16)));

#define BM 128
#define BN 128
#define KT1 32  // K-tile kernel 1 (one bf16 WMMA k-step)
#define S1  40  // bf16 LDS row stride for 32-k tiles: 80B rows (16B multiple)
#define KT2 64  // K-tile kernel 2 (two bf16 WMMA k-steps)
#define S2  72  // bf16 LDS row stride for 64-k tiles: 144B rows (16B multiple)

#define SHUF16(lo, hi) __builtin_shufflevector((lo), (hi), 0, 1, 2, 3, 4, 5, \
                                               6, 7, 8, 9, 10, 11, 12, 13, 14, 15)

struct HL { __bf16 h; __bf16 l; };

__device__ __forceinline__ HL split2(float x) {
    HL r;
    r.h = (__bf16)x;
    r.l = (__bf16)(x - (float)r.h);
    return r;
}

// ============================ Kernel 1 =====================================
__global__ __launch_bounds__(256)
void qlin_gemm1_quant(const float* __restrict__ X,      // [N, Din] row-major
                      const float* __restrict__ U,      // [Din, R] row-major
                      const float* __restrict__ lambda_x,
                      const float* __restrict__ lambda_w,
                      const float* __restrict__ codebook,   // [16]
                      const float* __restrict__ delta_x,
                      const float* __restrict__ rcodebook,  // [16]
                      __bf16* __restrict__ Abf,         // [N, R] out
                      int Din, int R)
{
    __shared__ __align__(16) __bf16 sxh[2][BM * S1];  // X tile hi [m][k]
    __shared__ __align__(16) __bf16 sxl[2][BM * S1];  // X tile lo [m][k]
    __shared__ __align__(16) __bf16 suh[2][BN * S1];  // U tile hi, transposed [n][k]
    __shared__ __align__(16) __bf16 sul[2][BN * S1];  // U tile lo, transposed [n][k]

    const int t     = threadIdx.x;
    const int lane  = t & 31;
    const int w     = t >> 5;
    const int mw    = (w >> 2) * 64;   // wave row origin in block tile
    const int nw    = (w & 3) * 32;    // wave col origin in block tile
    const int lhalf = lane >> 4;       // 0/1
    const int lmod  = lane & 15;
    const int m0    = blockIdx.y * BM;
    const int n0    = blockIdx.x * BN;

    v8f acc[4][2];
    const v8f vzero = {0.f, 0.f, 0.f, 0.f, 0.f, 0.f, 0.f, 0.f};
#pragma unroll
    for (int mt = 0; mt < 4; ++mt)
#pragma unroll
        for (int nt = 0; nt < 2; ++nt) acc[mt][nt] = vzero;

    const int nst = Din / KT1;
    float4 rx[4], ru[4];

    auto load_global = [&](int k0) {
#pragma unroll
        for (int i = 0; i < 4; ++i) {         // X: 1024 float4 (128 x 32)
            const int idx = t + i * 256;
            const int m   = idx >> 3;
            const int c   = idx & 7;
            rx[i] = *(const float4*)(X + (size_t)(m0 + m) * Din + k0 + c * 4);
        }
#pragma unroll
        for (int i = 0; i < 4; ++i) {         // U: 1024 float4 (32 x 128)
            const int idx = t + i * 256;
            const int kk  = idx >> 5;
            const int c   = idx & 31;
            ru[i] = *(const float4*)(U + (size_t)(k0 + kk) * R + n0 + c * 4);
        }
    };

    auto store_lds = [&](int b) {
#pragma unroll
        for (int i = 0; i < 4; ++i) {         // X hi/lo: b64 stores
            const int idx = t + i * 256;
            const int m   = idx >> 3;
            const int c   = idx & 7;
            const float4 v = rx[i];
            const HL e0 = split2(v.x), e1 = split2(v.y);
            const HL e2 = split2(v.z), e3 = split2(v.w);
            bf16x4 h, l;
            h[0] = e0.h; h[1] = e1.h; h[2] = e2.h; h[3] = e3.h;
            l[0] = e0.l; l[1] = e1.l; l[2] = e2.l; l[3] = e3.l;
            *(bf16x4*)&sxh[b][m * S1 + c * 4] = h;
            *(bf16x4*)&sxl[b][m * S1 + c * 4] = l;
        }
#pragma unroll
        for (int i = 0; i < 4; ++i) {         // U hi/lo: transposed scatter
            const int idx = t + i * 256;
            const int kk  = idx >> 5;
            const int c   = idx & 31;
            const float4 v = ru[i];
            const float vv[4] = {v.x, v.y, v.z, v.w};
#pragma unroll
            for (int j = 0; j < 4; ++j) {
                const HL e = split2(vv[j]);
                suh[b][(c * 4 + j) * S1 + kk] = e.h;
                sul[b][(c * 4 + j) * S1 + kk] = e.l;
            }
        }
    };

    load_global(0);
    store_lds(0);
    int buf = 0;

    for (int s = 0; s < nst; ++s) {
        __syncthreads();
        if (s + 1 < nst) load_global((s + 1) * KT1);
        if (s + 2 < nst) {                    // global_prefetch_b8 for s+2
            const int k2 = (s + 2) * KT1;
            __builtin_prefetch(X + (size_t)(m0 + (t >> 3)) * Din + k2 + (t & 7) * 4, 0, 0);
            __builtin_prefetch(U + (size_t)(k2 + (t >> 5)) * R + n0 + (t & 31) * 4, 0, 0);
        }

        // B fragments (k = (v%8) + (v/8)*16 + 8*(lane/16))
        bf16x16 bh[2], bl[2];
        const int kb = lhalf * 8;
#pragma unroll
        for (int nt = 0; nt < 2; ++nt) {
            const int n = nw + nt * 16 + lmod;
            bh[nt] = SHUF16(*(const bf16x8*)&suh[buf][n * S1 + kb],
                            *(const bf16x8*)&suh[buf][n * S1 + kb + 16]);
            bl[nt] = SHUF16(*(const bf16x8*)&sul[buf][n * S1 + kb],
                            *(const bf16x8*)&sul[buf][n * S1 + kb + 16]);
        }
        // A fragments (k = (v%8) + (v/8)*16 + 8*(lane/16)) + 3-product WMMA
        const int ka = lhalf * 8;
#pragma unroll
        for (int mt = 0; mt < 4; ++mt) {
            const int m = mw + mt * 16 + lmod;
            const bf16x16 ah = SHUF16(*(const bf16x8*)&sxh[buf][m * S1 + ka],
                                      *(const bf16x8*)&sxh[buf][m * S1 + ka + 16]);
            const bf16x16 al = SHUF16(*(const bf16x8*)&sxl[buf][m * S1 + ka],
                                      *(const bf16x8*)&sxl[buf][m * S1 + ka + 16]);
#pragma unroll
            for (int nt = 0; nt < 2; ++nt) {
                acc[mt][nt] = __builtin_amdgcn_wmma_f32_16x16x32_bf16(
                    false, ah, false, bh[nt], (short)0, acc[mt][nt], false, false);
                acc[mt][nt] = __builtin_amdgcn_wmma_f32_16x16x32_bf16(
                    false, ah, false, bl[nt], (short)0, acc[mt][nt], false, false);
                acc[mt][nt] = __builtin_amdgcn_wmma_f32_16x16x32_bf16(
                    false, al, false, bh[nt], (short)0, acc[mt][nt], false, false);
            }
        }
        __syncthreads();
        if (s + 1 < nst) { store_lds(buf ^ 1); buf ^= 1; }
    }

    // --- fused two-level quantization epilogue ---
    float cb[16], rcb[16];
#pragma unroll
    for (int i = 0; i < 16; ++i) { cb[i] = codebook[i]; rcb[i] = rcodebook[i]; }

#pragma unroll
    for (int nt = 0; nt < 2; ++nt) {
        const int r = n0 + nw + nt * 16 + lmod;
        float lx = lambda_x[r];
        float dx = delta_x[r];
        const float lw = lambda_w[r];
        if (fabsf(lx) < 1e-8f) lx = 1e-8f;
        if (fabsf(dx) < 1e-8f) dx = 1e-8f;
        const float inv_lx = 1.0f / lx;
        const float inv_dx = 1.0f / dx;
#pragma unroll
        for (int mt = 0; mt < 4; ++mt) {
#pragma unroll
            for (int vv = 0; vv < 8; ++vv) {
                const int m = m0 + mw + mt * 16 + vv + 8 * lhalf;
                const float p = acc[mt][nt][vv];
                const float z = p * inv_lx;   // nearest codebook (first wins ties)
                float zq = cb[0], bd = fabsf(z - cb[0]);
#pragma unroll
                for (int c = 1; c < 16; ++c) {
                    const float d = fabsf(z - cb[c]);
                    if (d < bd) { bd = d; zq = cb[c]; }
                }
                const float res = p - zq * lx;
                const float zr  = res * inv_dx;
                float rq = rcb[0], bd2 = fabsf(zr - rcb[0]);
#pragma unroll
                for (int c = 1; c < 16; ++c) {
                    const float d = fabsf(zr - rcb[c]);
                    if (d < bd2) { bd2 = d; rq = rcb[c]; }
                }
                const float aval = (zq * lx + rq * dx) * lw;
                Abf[(size_t)m * R + r] = (__bf16)aval;
            }
        }
    }
}

// ============================ Kernel 2 =====================================
__global__ __launch_bounds__(256)
void qlin_gemm2(const __bf16* __restrict__ Abf,   // [N, R]
                const float*  __restrict__ Zw,    // [R, Dout]
                const float*  __restrict__ bias,  // [Dout]
                float* __restrict__ Out,          // [N, Dout]
                int R, int Dout)
{
    __shared__ __align__(16) __bf16 sa[2][BM * S2];  // A tile [m][k]
    __shared__ __align__(16) __bf16 sb[2][BN * S2];  // Zw tile bf16, transposed [n][k]

    const int t     = threadIdx.x;
    const int lane  = t & 31;
    const int w     = t >> 5;
    const int mw    = (w >> 2) * 64;
    const int nw    = (w & 3) * 32;
    const int lhalf = lane >> 4;
    const int lmod  = lane & 15;
    const int m0    = blockIdx.y * BM;
    const int n0    = blockIdx.x * BN;

    v8f acc[4][2];
    const v8f vzero = {0.f, 0.f, 0.f, 0.f, 0.f, 0.f, 0.f, 0.f};
#pragma unroll
    for (int mt = 0; mt < 4; ++mt)
#pragma unroll
        for (int nt = 0; nt < 2; ++nt) acc[mt][nt] = vzero;

    const int nst = R / KT2;
    bf16x8 ra[4];
    float4 rb[8];

    auto load_global = [&](int k0) {
#pragma unroll
        for (int i = 0; i < 4; ++i) {         // A: 1024 chunks of 8 bf16
            const int idx = t + i * 256;
            const int m   = idx >> 3;
            const int c   = idx & 7;
            ra[i] = *(const bf16x8*)(Abf + (size_t)(m0 + m) * R + k0 + c * 8);
        }
#pragma unroll
        for (int i = 0; i < 8; ++i) {         // Zw: 2048 float4 (64 x 128)
            const int idx = t + i * 256;
            const int kk  = idx >> 5;
            const int c   = idx & 31;
            rb[i] = *(const float4*)(Zw + (size_t)(k0 + kk) * Dout + n0 + c * 4);
        }
    };

    auto store_lds = [&](int b) {
#pragma unroll
        for (int i = 0; i < 4; ++i) {         // A: b128 stores
            const int idx = t + i * 256;
            const int m   = idx >> 3;
            const int c   = idx & 7;
            *(bf16x8*)&sa[b][m * S2 + c * 8] = ra[i];
        }
#pragma unroll
        for (int i = 0; i < 8; ++i) {         // Zw: bf16 convert + transposed scatter
            const int idx = t + i * 256;
            const int kk  = idx >> 5;
            const int c   = idx & 31;
            const float4 v = rb[i];
            sb[b][(c * 4 + 0) * S2 + kk] = (__bf16)v.x;
            sb[b][(c * 4 + 1) * S2 + kk] = (__bf16)v.y;
            sb[b][(c * 4 + 2) * S2 + kk] = (__bf16)v.z;
            sb[b][(c * 4 + 3) * S2 + kk] = (__bf16)v.w;
        }
    };

    load_global(0);
    store_lds(0);
    int buf = 0;

    for (int s = 0; s < nst; ++s) {
        __syncthreads();
        if (s + 1 < nst) load_global((s + 1) * KT2);
        if (s + 2 < nst) {
            const int k2 = (s + 2) * KT2;
            __builtin_prefetch(Abf + (size_t)(m0 + (t >> 3)) * R + k2 + (t & 7) * 8, 0, 0);
            __builtin_prefetch(Zw + (size_t)(k2 + (t >> 5)) * Dout + n0 + (t & 31) * 4, 0, 0);
        }

#pragma unroll
        for (int kk = 0; kk < KT2; kk += 32) {
            bf16x16 bfrag[2];
            const int kb = kk + lhalf * 8;
#pragma unroll
            for (int nt = 0; nt < 2; ++nt) {
                const int n = nw + nt * 16 + lmod;
                bfrag[nt] = SHUF16(*(const bf16x8*)&sb[buf][n * S2 + kb],
                                   *(const bf16x8*)&sb[buf][n * S2 + kb + 16]);
            }
            const int ka = kk + lhalf * 8;
#pragma unroll
            for (int mt = 0; mt < 4; ++mt) {
                const int m = mw + mt * 16 + lmod;
                const bf16x16 afrag =
                    SHUF16(*(const bf16x8*)&sa[buf][m * S2 + ka],
                           *(const bf16x8*)&sa[buf][m * S2 + ka + 16]);
#pragma unroll
                for (int nt = 0; nt < 2; ++nt)
                    acc[mt][nt] = __builtin_amdgcn_wmma_f32_16x16x32_bf16(
                        false, afrag, false, bfrag[nt], (short)0, acc[mt][nt],
                        false, false);
            }
        }
        __syncthreads();
        if (s + 1 < nst) { store_lds(buf ^ 1); buf ^= 1; }
    }

    // epilogue: + bias, write f32
#pragma unroll
    for (int nt = 0; nt < 2; ++nt) {
        const int n  = n0 + nw + nt * 16 + lmod;
        const float bz = bias[n];
#pragma unroll
        for (int mt = 0; mt < 4; ++mt) {
#pragma unroll
            for (int vv = 0; vv < 8; ++vv) {
                const int m = m0 + mw + mt * 16 + vv + 8 * lhalf;
                Out[(size_t)m * Dout + n] = acc[mt][nt][vv] + bz;
            }
        }
    }
}

// ============================ Launch =======================================
extern "C" void kernel_launch(void* const* d_in, const int* in_sizes, int n_in,
                              void* d_out, int out_size, void* d_ws, size_t ws_size,
                              hipStream_t stream) {
    const float* X         = (const float*)d_in[0];  // hidden_states [1,S,Din]
    const float* U         = (const float*)d_in[1];  // [Din, R]
    const float* lambda_x  = (const float*)d_in[2];  // [R]
    const float* lambda_w  = (const float*)d_in[3];  // [R]
    const float* Zw        = (const float*)d_in[4];  // [R, Dout]
    const float* codebook  = (const float*)d_in[5];  // [16]
    const float* delta_x   = (const float*)d_in[6];  // [R]
    const float* rcodebook = (const float*)d_in[7];  // [16]
    const float* bias      = (const float*)d_in[8];  // [Dout]
    float* out = (float*)d_out;

    const int R    = in_sizes[2];
    const int Din  = in_sizes[1] / R;
    const int Ntok = in_sizes[0] / Din;   // 4096
    const int Dout = in_sizes[8];

    __bf16* Abf = (__bf16*)d_ws;          // bf16 intermediate [Ntok, R] = 16 MB

    dim3 blk(256);
    dim3 g1(R / BN, Ntok / BM);
    dim3 g2(Dout / BN, Ntok / BM);
    qlin_gemm1_quant<<<g1, blk, 0, stream>>>(X, U, lambda_x, lambda_w, codebook,
                                             delta_x, rcodebook, Abf, Din, R);
    qlin_gemm2<<<g2, blk, 0, stream>>>(Abf, Zw, bias, out, R, Dout);
}
